// AttnLSTMEmbedding_74208444940934
// MI455X (gfx1250) — matrix operand order repaired and checked by reading, not compiled, not run on hardware
//
#include <hip/hip_runtime.h>
#include <math.h>

// ---------------------------------------------------------------------------
// AttnLSTM embedding for MI455X (gfx1250), bf16 WMMA path.
// Shapes: n_test=512, n_support=2048, n_feat=2048, W:(4096,8192), U:(2048,8192)
// z-GEMM dominates: fuse [q,r]@W + h@U into [q,r,h](512x6144) @ [W;U](6144x8192)
// Weights converted once to bf16 (96MB) -> resident in 192MB L2 across the
// 10 sequential iterations; GEMMs use v_wmma_f32_16x16x32_bf16 (f32 accum).
// GEMM: block tile 64(M)x256(N), 8 waves in 2x4, wave tile 32x64 (8 acc),
// K-step 32; LDS rows padded to 40 u16 (80B). A-tile (and B-tile when no
// transpose is needed) staged with GLOBAL_LOAD_ASYNC_TO_LDS_B128 (ASYNCcnt),
// waited on with s_wait_asynccnt 0 before the barrier that precedes the WMMAs.
// ---------------------------------------------------------------------------

typedef __attribute__((ext_vector_type(16))) __bf16 v16bf;
typedef __attribute__((ext_vector_type(8)))  float  v8f;
typedef __attribute__((ext_vector_type(4)))  unsigned int u32x4;

#define N_TEST  512
#define N_SUP   2048
#define N_FEAT  2048
#define K_FUSED 6144   // 2*n_feat (y=[q,r]) + n_feat (h)
#define N_GATE  8192   // 4*n_feat

__device__ __forceinline__ unsigned short f32_to_bf16(float f) {
    union { float f; unsigned u; } x; x.f = f;
    unsigned r = x.u + 0x7FFFu + ((x.u >> 16) & 1u);   // round-to-nearest-even
    return (unsigned short)(r >> 16);
}

__device__ __forceinline__ float hsig(float v) {
    return fminf(fmaxf(0.2f * v + 0.5f, 0.0f), 1.0f);
}

// Async global -> LDS 128-bit copy (CDNA5; tracked by ASYNCcnt).
// lds_off: per-lane LDS byte offset (low 32 bits of a __shared__ pointer).
__device__ __forceinline__ void async_copy_b128(unsigned lds_off,
                                                unsigned long long gaddr) {
    asm volatile("global_load_async_to_lds_b128 %0, %1, off"
                 :: "v"(lds_off), "v"(gaddr) : "memory");
}
__device__ __forceinline__ void wait_async0() {
    asm volatile("s_wait_asynccnt 0x0" ::: "memory");
}

__device__ __forceinline__ float block_reduce_sum(float v, float* red) {
    int tid = threadIdx.x;
    red[tid] = v; __syncthreads();
    for (int off = 128; off > 0; off >>= 1) {
        if (tid < off) red[tid] += red[tid + off];
        __syncthreads();
    }
    float r = red[0]; __syncthreads();
    return r;
}

__device__ __forceinline__ float block_reduce_max(float v, float* red) {
    int tid = threadIdx.x;
    red[tid] = v; __syncthreads();
    for (int off = 128; off > 0; off >>= 1) {
        if (tid < off) red[tid] = fmaxf(red[tid], red[tid + off]);
        __syncthreads();
    }
    float r = red[0]; __syncthreads();
    return r;
}

// ---------------------------------------------------------------------------
// bf16 GEMM: C(MxN) = A(MxK) * B(KxN), f32 accumulation via WMMA.
//   BT=1       : B element (k,n) at B[n*ldb + k]  (e = xq @ xp^T)
//   OUT_BF16=1 : write C as bf16 (into the fused [q,r,h] activation buffer)
// ---------------------------------------------------------------------------
template<int BT, int OUT_BF16>
__global__ __launch_bounds__(256)
void gemm_bf16(const unsigned short* __restrict__ A, int lda,
               const unsigned short* __restrict__ B, int ldb,
               void* __restrict__ C, int ldc, int K)
{
    __shared__ unsigned short As[64][40];    //  5.0 KB, [m][k]
    __shared__ unsigned short Bs[256][40];   // 20.0 KB, [n][k] (k-contiguous)

    const int tid  = threadIdx.x;
    const int m0   = blockIdx.y * 64;
    const int n0   = blockIdx.x * 256;
    const int wave = tid >> 5;
    const int lane = tid & 31;
    const int wm   = (wave >> 2) * 32;       // 0 | 32
    const int wn   = (wave & 3) * 64;        // 0,64,128,192
    const int half = lane >> 4;              // lane/16
    const int lm   = lane & 15;

    v8f acc[2][4] = {};

    // staging coordinates (hoisted)
    const int am  = tid >> 2;                // 0..63
    const int akq = (tid & 3) * 8;           // 0,8,16,24
    const int bn  = tid;                     // BT=1: 0..255
    const int bkp = (tid >> 4) * 2;          // BT=0: 0..30
    const int bn16 = (tid & 15) * 16;        // BT=0: 0..240

    const unsigned ldsA = (unsigned)(size_t)&As[am][akq];          // 16B aligned
    const unsigned ldsB = (unsigned)(size_t)&Bs[bn][0];            // 16B aligned

    for (int k0 = 0; k0 < K; k0 += 32) {
        // --- stage A tile 64x32: one async b128 per thread ---
        async_copy_b128(ldsA,
            (unsigned long long)(size_t)(A + (size_t)(m0 + am) * lda + (k0 + akq)));
        // --- stage B tile 32(k) x 256(n) into Bs[n][k] ---
        if (BT) {
            unsigned long long gb =
                (unsigned long long)(size_t)(B + (size_t)(n0 + bn) * ldb + k0);
            #pragma unroll
            for (int q = 0; q < 4; ++q)
                async_copy_b128(ldsB + (unsigned)(q * 16), gb + (unsigned long long)(q * 16));
        } else {
            const unsigned short* b0 = B + (size_t)(k0 + bkp) * ldb + (n0 + bn16);
            const unsigned short* b1 = b0 + ldb;
            union { u32x4 v; unsigned short s[8]; } r0a, r0b, r1a, r1b;
            r0a.v = *(const u32x4*)b0;  r0b.v = *(const u32x4*)(b0 + 8);
            r1a.v = *(const u32x4*)b1;  r1b.v = *(const u32x4*)(b1 + 8);
            #pragma unroll
            for (int j = 0; j < 8; ++j) {
                *(unsigned*)&Bs[bn16 + j][bkp] =
                    (unsigned)r0a.s[j] | ((unsigned)r1a.s[j] << 16);
                *(unsigned*)&Bs[bn16 + 8 + j][bkp] =
                    (unsigned)r0b.s[j] | ((unsigned)r1b.s[j] << 16);
            }
            if (k0 + 32 < K) __builtin_prefetch(b0 + 32 * (size_t)ldb, 0, 1);
        }
        wait_async0();          // async LDS writes visible before the barrier
        __syncthreads();

        // --- assemble fragments ---
        union Frag { v16bf v; unsigned u[8]; };
        Frag af[2], bfm[4];
        #pragma unroll
        for (int i = 0; i < 2; ++i) {
            #pragma unroll
            for (int j = 0; j < 8; ++j) {
                // A 16x32 bf16 layout: vgpr j<4 -> K = 8*half+2j ; j>=4 -> 16+8*half+2(j-4)
                int ka = (j < 4) ? (8 * half + 2 * j) : (16 + 8 * half + 2 * (j - 4));
                af[i].u[j] = *(const unsigned*)&As[wm + i * 16 + lm][ka];
            }
        }
        #pragma unroll
        for (int bi = 0; bi < 4; ++bi) {
            #pragma unroll
            for (int j = 0; j < 8; ++j) {
                // B 32x16 bf16 layout: vgpr j<4 -> K = 8*half+2j ; j>=4 -> 16+8*half+2(j-4)
                int kb = (j < 4) ? (8 * half + 2 * j) : (16 + 8 * half + 2 * (j - 4));
                bfm[bi].u[j] = *(const unsigned*)&Bs[wn + bi * 16 + lm][kb];
            }
        }
        // --- 8 independent WMMAs per stage ---
        #pragma unroll
        for (int i = 0; i < 2; ++i)
            #pragma unroll
            for (int bi = 0; bi < 4; ++bi)
                acc[i][bi] = __builtin_amdgcn_wmma_f32_16x16x32_bf16(
                    false, af[i].v, false, bfm[bi].v, (short)0, acc[i][bi],
                    false, false);
        __syncthreads();
    }

    // --- store C: vgpr v -> row m = v + 8*half, col n = lm ---
    #pragma unroll
    for (int i = 0; i < 2; ++i) {
        #pragma unroll
        for (int bi = 0; bi < 4; ++bi) {
            #pragma unroll
            for (int v = 0; v < 8; ++v) {
                int m = m0 + wm + i * 16 + v + 8 * half;
                int n = n0 + wn + bi * 16 + lm;
                if (OUT_BF16)
                    ((unsigned short*)C)[(size_t)m * ldc + n] = f32_to_bf16(acc[i][bi][v]);
                else
                    ((float*)C)[(size_t)m * ldc + n] = acc[i][bi][v];
            }
        }
    }
}

// --- one-time: [W;U] -> bf16 (6144 x 8192), 8 elements per thread ----------
__global__ __launch_bounds__(256)
void convert_wu(const float* __restrict__ W, const float* __restrict__ U,
                unsigned short* __restrict__ WU)
{
    size_t base = ((size_t)blockIdx.x * 256 + threadIdx.x) * 8;
    #pragma unroll
    for (int j = 0; j < 8; ++j) {
        size_t idx = base + j;
        int row = (int)(idx >> 13);          // /8192
        int col = (int)(idx & 8191);
        float v = (row < 4096) ? W[(size_t)row * 8192 + col]
                               : U[(size_t)(row - 4096) * 8192 + col];
        WU[idx] = f32_to_bf16(v);
    }
}

// --- one-time: xp -> bf16 + per-block partial sum of xp^2 ------------------
__global__ __launch_bounds__(256)
void convert_xp(const float* __restrict__ xp, unsigned short* __restrict__ xpbf,
                float* __restrict__ partials)
{
    __shared__ float red[256];
    size_t base = ((size_t)blockIdx.x * 256 + threadIdx.x) * 8;
    float s = 0.f;
    #pragma unroll
    for (int j = 0; j < 8; ++j) {
        float v = xp[base + j];
        xpbf[base + j] = f32_to_bf16(v);
        s += v * v;
    }
    float tot = block_reduce_sum(s, red);
    if (threadIdx.x == 0) partials[blockIdx.x] = tot;
}

// --- init: q = q_init, c = 0, h(bf16 region of fused A) = 0 ----------------
__global__ __launch_bounds__(256)
void init_state(const float* __restrict__ q_init, float* __restrict__ q,
                float* __restrict__ c, unsigned short* __restrict__ ybuf)
{
    size_t idx = (size_t)blockIdx.x * 256 + threadIdx.x;   // 512*2048 threads
    int row = (int)(idx >> 11), col = (int)(idx & 2047);
    q[idx] = q_init[idx];
    c[idx] = 0.f;
    ybuf[(size_t)row * K_FUSED + 2 * N_FEAT + col] = 0;    // bf16 zero
}

// --- per-iter: xq = x + q (bf16), q -> fused-A left block, Sum(xq^2) -------
__global__ __launch_bounds__(256)
void stage_xq(const float* __restrict__ x, const float* __restrict__ q,
              unsigned short* __restrict__ xqbf, unsigned short* __restrict__ ybuf,
              float* __restrict__ partials)
{
    __shared__ float red[256];
    size_t base = ((size_t)blockIdx.x * 256 + threadIdx.x) * 8;
    float s = 0.f;
    #pragma unroll
    for (int j = 0; j < 8; ++j) {
        size_t idx = base + j;
        float qv = q[idx];
        float xq = x[idx] + qv;
        xqbf[idx] = f32_to_bf16(xq);
        int row = (int)(idx >> 11), col = (int)(idx & 2047);
        ybuf[(size_t)row * K_FUSED + col] = f32_to_bf16(qv);
        s += xq * xq;
    }
    float tot = block_reduce_sum(s, red);
    if (threadIdx.x == 0) partials[blockIdx.x] = tot;
}

// --- per-iter: denom + row softmax of e/denom -> a (bf16). 1 block/row -----
__global__ __launch_bounds__(256)
void softmax_row(const float* __restrict__ e, unsigned short* __restrict__ abf,
                 const float* __restrict__ p_xq, int n_xq,
                 const float* __restrict__ p_xp, int n_xp)
{
    __shared__ float red[256];
    int row = blockIdx.x, tid = threadIdx.x;

    float s = 0.f;
    for (int i = tid; i < n_xq; i += 256) s += p_xq[i];
    float sum_xq = block_reduce_sum(s, red);
    s = 0.f;
    for (int i = tid; i < n_xp; i += 256) s += p_xp[i];
    float sum_xp = block_reduce_sum(s, red);
    float inv_denom = 1.0f / (sqrtf(sum_xq * sum_xp) + 1e-7f);

    const float* er = e + (size_t)row * N_SUP;
    float ev[8];
    float m = -3.402823466e+38f;
    #pragma unroll
    for (int j = 0; j < 8; ++j) {
        ev[j] = er[tid + 256 * j] * inv_denom;
        m = fmaxf(m, ev[j]);
    }
    m = block_reduce_max(m, red);

    float lsum = 0.f;
    #pragma unroll
    for (int j = 0; j < 8; ++j) {
        ev[j] = __expf(ev[j] - m);
        lsum += ev[j];
    }
    float tot = block_reduce_sum(lsum, red);
    float inv = 1.0f / tot;
    unsigned short* ar = abf + (size_t)row * N_SUP;
    #pragma unroll
    for (int j = 0; j < 8; ++j)
        ar[tid + 256 * j] = f32_to_bf16(ev[j] * inv);
}

// --- per-iter: bias + LSTM gates; writes c, q(=o), h(bf16 into fused A) ----
__global__ __launch_bounds__(256)
void gates(const float* __restrict__ z, const float* __restrict__ b,
           float* __restrict__ c, float* __restrict__ q,
           unsigned short* __restrict__ ybuf)
{
    size_t idx = (size_t)blockIdx.x * 256 + threadIdx.x;   // 512*2048 threads
    int t = (int)(idx >> 11), f = (int)(idx & 2047);
    const float* zr = z + (size_t)t * N_GATE;
    float z0 = zr[f]                + b[f];
    float z1 = zr[f + N_FEAT]       + b[f + N_FEAT];
    float z2 = zr[f + 2 * N_FEAT]   + b[f + 2 * N_FEAT];
    float z3 = zr[f + 3 * N_FEAT]   + b[f + 3 * N_FEAT];
    float ig = hsig(z0), fg = hsig(z1), og = hsig(z3);
    float cn = fg * c[idx] + ig * tanhf(z2);
    float hn = og * tanhf(cn);
    c[idx] = cn;
    q[idx] = og;                                            // next q = o
    ybuf[(size_t)t * K_FUSED + 2 * N_FEAT + f] = f32_to_bf16(hn);
}

// --- finalize: out = concat(x + q, xp) -------------------------------------
__global__ __launch_bounds__(256)
void finalize(const float* __restrict__ x, const float* __restrict__ q,
              const float* __restrict__ xp, float* __restrict__ out)
{
    size_t idx = (size_t)blockIdx.x * 256 + threadIdx.x;
    const size_t n0 = (size_t)N_TEST * N_FEAT;              // 1,048,576
    const size_t n1 = n0 + (size_t)N_SUP * N_FEAT;          // 5,242,880
    if (idx < n0)       out[idx] = x[idx] + q[idx];
    else if (idx < n1)  out[idx] = xp[idx - n0];
}

// ---------------------------------------------------------------------------
extern "C" void kernel_launch(void* const* d_in, const int* in_sizes, int n_in,
                              void* d_out, int out_size, void* d_ws, size_t ws_size,
                              hipStream_t stream)
{
    (void)in_sizes; (void)n_in; (void)out_size; (void)ws_size;
    const float* x      = (const float*)d_in[0];
    const float* xp     = (const float*)d_in[1];
    const float* q_init = (const float*)d_in[2];
    const float* W      = (const float*)d_in[3];
    const float* U      = (const float*)d_in[4];
    const float* b      = (const float*)d_in[5];
    float* out          = (float*)d_out;

    char* ws = (char*)d_ws;
    unsigned short* WUbf = (unsigned short*)ws; ws += (size_t)K_FUSED * N_GATE * 2;  // 96 MB
    unsigned short* xpbf = (unsigned short*)ws; ws += (size_t)N_SUP  * N_FEAT * 2;   //  8 MB
    unsigned short* ybuf = (unsigned short*)ws; ws += (size_t)N_TEST * K_FUSED * 2;  //  6 MB [q,r,h]
    unsigned short* xqbf = (unsigned short*)ws; ws += (size_t)N_TEST * N_FEAT * 2;   //  2 MB
    unsigned short* abf  = (unsigned short*)ws; ws += (size_t)N_TEST * N_SUP  * 2;   //  2 MB
    float* ebuf = (float*)ws; ws += (size_t)N_TEST * N_SUP  * 4;                     //  4 MB
    float* zbuf = (float*)ws; ws += (size_t)N_TEST * N_GATE * 4;                     // 16 MB
    float* qbuf = (float*)ws; ws += (size_t)N_TEST * N_FEAT * 4;                     //  4 MB
    float* cbuf = (float*)ws; ws += (size_t)N_TEST * N_FEAT * 4;                     //  4 MB
    float* p_xp = (float*)ws; ws += 2048 * 4;
    float* p_xq = (float*)ws; ws += 512 * 4;

    convert_wu<<<24576, 256, 0, stream>>>(W, U, WUbf);
    convert_xp<<<2048, 256, 0, stream>>>(xp, xpbf, p_xp);
    init_state<<<4096, 256, 0, stream>>>(q_init, qbuf, cbuf, ybuf);

    for (int it = 0; it < 10; ++it) {
        stage_xq<<<512, 256, 0, stream>>>(x, qbuf, xqbf, ybuf, p_xq);
        // e = xq @ xp^T   (M=512, N=2048, K=2048), B transposed
        gemm_bf16<1, 0><<<dim3(8, 8), 256, 0, stream>>>(
            xqbf, N_FEAT, xpbf, N_FEAT, ebuf, N_SUP, N_FEAT);
        softmax_row<<<512, 256, 0, stream>>>(ebuf, abf, p_xq, 512, p_xp, 2048);
        // r = a @ xp      (M=512, N=2048, K=2048), bf16 result into ybuf[:,2048:4096]
        gemm_bf16<0, 1><<<dim3(8, 8), 256, 0, stream>>>(
            abf, N_SUP, xpbf, N_FEAT, (void*)(ybuf + N_FEAT), K_FUSED, N_SUP);
        // z = [q,r,h] @ [W;U]  (M=512, N=8192, K=6144)
        gemm_bf16<0, 0><<<dim3(32, 8), 256, 0, stream>>>(
            ybuf, K_FUSED, WUbf, N_GATE, zbuf, N_GATE, K_FUSED);
        gates<<<4096, 256, 0, stream>>>(zbuf, b, cbuf, qbuf, ybuf);
    }
    finalize<<<20480, 256, 0, stream>>>(x, qbuf, xp, out);
}
